// MiddleEncoder_9268539425522
// MI455X (gfx1250) — matrix operand order repaired and checked by reading, not compiled, not run on hardware
//
#include <hip/hip_runtime.h>
#include <hip/hip_bf16.h>

// ---------------------------------------------------------------------------
// CDNA5 (gfx1250) sparse-conv stack:
//   4x (gather -> per-offset 32x32 GEMM via V_WMMA_F32_16X16X32_BF16 -> scatter
//       atomic-add) + bias/ReLU, residual merge, 32->20 logits GEMM.
// Feature maps (19.2 MB) are L2-resident on MI455X (192 MB L2); weights are
// staged into LDS with the gfx1250 async global->LDS copy; scatter uses
// clause'd saddr-form global_atomic_add_f32.
// ---------------------------------------------------------------------------

typedef __attribute__((ext_vector_type(16))) __bf16 v16bf;
typedef __attribute__((ext_vector_type(8)))  float  v8f;

union ABu {
    v16bf v;
    uint4 q[2];
};

__device__ __forceinline__ unsigned short f2bf(float f) {
    unsigned u = __float_as_uint(f);
    u += 0x7FFFu + ((u >> 16) & 1u);      // round-to-nearest-even
    return (unsigned short)(u >> 16);
}

// ---------------------------------------------------------------------------
// Pack one weight set W[K][32][32] (f32, [kvol, Cin, Cout]) into a contiguous-K
// B-matrix VGPR layout for V_WMMA_F32_16X16X32_BF16:
//   lane l: column N = l & 15, K-half = (l >> 4) * 16,
//   element e (0..15): B[Khalf + e][N]   (bf16, 2 per dword, K-consecutive)
// Stored as wp[k][h][lane][e]  (h = Cout half, 16 bf16 = 32 B per lane).
// ---------------------------------------------------------------------------
__global__ __launch_bounds__(256)
void pack_w(const float* __restrict__ w, unsigned short* __restrict__ wp, int kvol) {
    int t = blockIdx.x * blockDim.x + threadIdx.x;
    if (t >= kvol * 2 * 32) return;
    int lane  = t & 31;
    int h     = (t >> 5) & 1;
    int k     = t >> 6;
    int ncol  = lane & 15;
    int kbase = (lane >> 4) * 16;
    unsigned short* dst = wp + (size_t)t * 16;   // t == (k*2+h)*32+lane
#pragma unroll
    for (int e = 0; e < 16; ++e) {
        float f = w[((size_t)k * 32 + (kbase + e)) * 32 + (h * 16 + ncol)];
        dst[e] = f2bf(f);
    }
}

// f32 -> bf16 activation convert (4-wide)
__global__ __launch_bounds__(256)
void cvt_bf16(const float* __restrict__ in, unsigned short* __restrict__ out, int total4) {
    int i = blockIdx.x * blockDim.x + threadIdx.x;
    if (i >= total4) return;
    float4 v = ((const float4*)in)[i];
    ushort4 o = { f2bf(v.x), f2bf(v.y), f2bf(v.z), f2bf(v.w) };
    ((ushort4*)out)[i] = o;
}

// acc + bias -> ReLU -> bf16 activations for the next layer (4-wide; C=32)
__global__ __launch_bounds__(256)
void bias_relu_bf16(const float* __restrict__ acc, const float* __restrict__ bias,
                    unsigned short* __restrict__ out, int total4) {
    int i = blockIdx.x * blockDim.x + threadIdx.x;
    if (i >= total4) return;
    float4 v = ((const float4*)acc)[i];
    int c0 = (i * 4) & 31;
    float a0 = v.x + bias[c0 + 0]; a0 = a0 > 0.f ? a0 : 0.f;
    float a1 = v.y + bias[c0 + 1]; a1 = a1 > 0.f ? a1 : 0.f;
    float a2 = v.z + bias[c0 + 2]; a2 = a2 > 0.f ? a2 : 0.f;
    float a3 = v.w + bias[c0 + 3]; a3 = a3 > 0.f ? a3 : 0.f;
    ushort4 o = { f2bf(a0), f2bf(a1), f2bf(a2), f2bf(a3) };
    ((ushort4*)out)[i] = o;
}

// ---------------------------------------------------------------------------
// One wave per (16-row tile, kernel offset k = blockIdx.y).
//   A (16x32 bf16): lane l holds row M = l&15; elems 0..15 = cols h16..h16+15
//                   with h16 = (l>>4)*16 -> 2x b128
//   B (32x16 bf16): prepacked in LDS (async global->LDS), 2x ds_load_b128
//   D (16x16 f32):  VGPR r: row M = 8*(l>>4)+r, col N = l&15
// Scatter-add via relaxed device-scope global_atomic_add_f32; the rare ragged
// tail tile takes a separate (wave-uniform) clamped index path that redirects
// out-of-range rows to a dummy accumulator row (row n).
// ---------------------------------------------------------------------------
__global__ __launch_bounds__(256)
void spconv16(const unsigned short* __restrict__ act,   // N x 32 bf16
              const int* __restrict__ gidx,             // K x N gather rows
              const int* __restrict__ sidx,             // K x N scatter rows
              const unsigned short* __restrict__ wp,    // packed weights
              float* __restrict__ acc,                  // (N+16) x 32 f32, zeroed
              int n) {
    __shared__ uint4 smem[128];                // W[k], both Cout halves: 2 KB

    const int t    = threadIdx.x;
    const int lane = t & 31;
    const int wave = t >> 5;
    const int k    = blockIdx.y;

    // ---- stage packed W[k] into LDS with gfx1250 async copy ----
    if (t < 128) {
        const uint4* src = (const uint4*)wp + (size_t)k * 128 + t;
        // Low 32 bits of a generic LDS pointer are the LDS byte offset
        // (aperture rule).
        unsigned ldsa = (unsigned)(size_t)&smem[t];
        asm volatile("global_load_async_to_lds_b128 %0, %1, off"
                     :: "v"(ldsa), "v"(src) : "memory");
    }
    asm volatile("s_wait_asynccnt 0x0" ::: "memory");
    __syncthreads();

    const int tile = blockIdx.x * 8 + wave;
    const int row0 = tile * 16;
    if (row0 >= n) return;                     // wave-uniform

    const int half  = lane >> 4;
    const int m     = lane & 15;
    const int ncol  = lane & 15;
    const int mbase = half * 8;

    const int* g = gidx + (size_t)k * n + row0;
    const int* s = sidx + (size_t)k * n + row0;

    unsigned grow;                 // gathered input row for this lane's M
    unsigned sofs[8];              // scatter row offsets (elements) for D rows
    if (row0 + 16 <= n) {
        // ---- fast path (all but at most one tile): no clamping ----
        grow = (unsigned)g[m];
        int4 sa = *(const int4*)(s + mbase);
        int4 sb = *(const int4*)(s + mbase + 4);
        sofs[0] = (unsigned)sa.x * 32u; sofs[1] = (unsigned)sa.y * 32u;
        sofs[2] = (unsigned)sa.z * 32u; sofs[3] = (unsigned)sa.w * 32u;
        sofs[4] = (unsigned)sb.x * 32u; sofs[5] = (unsigned)sb.y * 32u;
        sofs[6] = (unsigned)sb.z * 32u; sofs[7] = (unsigned)sb.w * 32u;
    } else {
        // ---- ragged tail tile: clamp reads, dump extra rows into row n ----
        int lim = n - row0;                    // 1..15 valid rows
        grow = (unsigned)g[m < lim ? m : lim - 1];
#pragma unroll
        for (int r = 0; r < 8; ++r) {
            int pos = mbase + r;
            unsigned sr = (pos < lim) ? (unsigned)s[pos] : (unsigned)n;
            sofs[r] = sr * 32u;
        }
    }

    // --- gather A row for this lane ---
    const unsigned abase = grow * 32u + (unsigned)(half * 16);
    ABu a;
    a.q[0] = *(const uint4*)(act + abase);        // cols h16 .. h16+7
    a.q[1] = *(const uint4*)(act + abase + 8u);   // cols h16+8 .. h16+15

#pragma unroll
    for (int h = 0; h < 2; ++h) {              // Cout halves
        const uint4* wb = &smem[(h * 32 + lane) * 2];
        ABu b;
        b.q[0] = wb[0];
        b.q[1] = wb[1];

        v8f c = {0.f, 0.f, 0.f, 0.f, 0.f, 0.f, 0.f, 0.f};
        c = __builtin_amdgcn_wmma_f32_16x16x32_bf16(
                /*neg_a=*/false, a.v, /*neg_b=*/false, b.v,
                /*c_mod=*/(short)0, c, /*reuse_a=*/false, /*reuse_b=*/false);

        const unsigned hc = (unsigned)(h * 16 + ncol);
#pragma unroll
        for (int r = 0; r < 8; ++r) {
            __hip_atomic_fetch_add(acc + (sofs[r] + hc), c[r],
                                   __ATOMIC_RELAXED, __HIP_MEMORY_SCOPE_AGENT);
        }
    }
}

// bias+ReLU both branches, residual add, 32x20 logits GEMM (f32 VALU; tiny)
__global__ __launch_bounds__(256)
void merge_logits(const float* __restrict__ accS, const float* __restrict__ bS,
                  const float* __restrict__ accR, const float* __restrict__ bR,
                  const float* __restrict__ wl,   // 32 x 20
                  const float* __restrict__ bl,   // 20
                  float* __restrict__ out, int n, int ncls) {
    int i = blockIdx.x * blockDim.x + threadIdx.x;
    if (i >= n) return;
    float f[32];
#pragma unroll
    for (int c = 0; c < 32; ++c) {
        float sv = accS[(size_t)i * 32 + c] + bS[c];
        float rv = accR[(size_t)i * 32 + c] + bR[c];
        f[c] = (sv > 0.0f ? sv : 0.0f) + (rv > 0.0f ? rv : 0.0f);
    }
    for (int j = 0; j < ncls; ++j) {
        float a = bl[j];
#pragma unroll
        for (int c = 0; c < 32; ++c) a = fmaf(f[c], wl[c * ncls + j], a);
        out[(size_t)i * ncls + j] = a;
    }
}

extern "C" void kernel_launch(void* const* d_in, const int* in_sizes, int n_in,
                              void* d_out, int out_size, void* d_ws, size_t ws_size,
                              hipStream_t stream) {
    const float* in_feats = (const float*)d_in[0];
    const int*   rb_g_a   = (const int*)d_in[1];
    const int*   rb_s_a   = (const int*)d_in[2];
    const int*   rb_g_b   = (const int*)d_in[3];
    const int*   rb_s_b   = (const int*)d_in[4];
    const float* w1   = (const float*)d_in[5];
    const float* b1   = (const float*)d_in[6];
    const float* w1_2 = (const float*)d_in[7];
    const float* b1_2 = (const float*)d_in[8];
    const float* w2   = (const float*)d_in[9];
    const float* b2   = (const float*)d_in[10];
    const float* w3   = (const float*)d_in[11];
    const float* b3   = (const float*)d_in[12];
    const float* wl   = (const float*)d_in[13];
    const float* bl   = (const float*)d_in[14];

    const int N  = in_sizes[0] / 32;           // 150000
    const int K  = in_sizes[1] / N;            // 27
    const int NC = in_sizes[14];               // 20 classes

    // ---- workspace carve-up (256B aligned) ----
    const size_t wpBytes  = (size_t)K * 2 * 32 * 16 * sizeof(unsigned short);   // 55 KB / set
    const size_t bfBytes  = (size_t)N * 32 * sizeof(unsigned short);            // 9.6 MB
    const size_t accBytes = (size_t)(N + 16) * 32 * sizeof(float);              // 19.2 MB (+dummy rows)
    char*  p   = (char*)d_ws;
    size_t off = 0;
    auto carve = [&](size_t bytes) { void* q = p + off; off = (off + bytes + 255) & ~(size_t)255; return q; };
    unsigned short* wp1  = (unsigned short*)carve(wpBytes);
    unsigned short* wp12 = (unsigned short*)carve(wpBytes);
    unsigned short* wp2  = (unsigned short*)carve(wpBytes);
    unsigned short* wp3  = (unsigned short*)carve(wpBytes);
    unsigned short* inbf = (unsigned short*)carve(bfBytes);
    unsigned short* actb = (unsigned short*)carve(bfBytes);
    float*          accS = (float*)carve(accBytes);
    float*          accT = (float*)carve(accBytes);
    (void)ws_size; (void)n_in; (void)out_size;

    // ---- pack weights into WMMA B layout ----
    {
        int threads = K * 2 * 32;
        int blocks  = (threads + 255) / 256;
        pack_w<<<blocks, 256, 0, stream>>>(w1,   wp1,  K);
        pack_w<<<blocks, 256, 0, stream>>>(w1_2, wp12, K);
        pack_w<<<blocks, 256, 0, stream>>>(w2,   wp2,  K);
        pack_w<<<blocks, 256, 0, stream>>>(w3,   wp3,  K);
    }

    const int total4 = (N * 32) / 4;
    const int eblk   = (total4 + 255) / 256;
    cvt_bf16<<<eblk, 256, 0, stream>>>(in_feats, inbf, total4);

    const int tiles = (N + 15) / 16;
    dim3 cgrid((tiles + 7) / 8, K);            // 8 waves (tiles) per 256-thread block

    // shortcut = relu(spconv(in, rb_a, w1) + b1)
    hipMemsetAsync(accT, 0, accBytes, stream);
    spconv16<<<cgrid, 256, 0, stream>>>(inbf, rb_g_a, rb_s_a, wp1, accT, N);
    bias_relu_bf16<<<eblk, 256, 0, stream>>>(accT, b1, actb, total4);

    // shortcut = relu(spconv(shortcut, rb_b, w1_2) + b1_2) -> raw acc in accS
    hipMemsetAsync(accS, 0, accBytes, stream);
    spconv16<<<cgrid, 256, 0, stream>>>(actb, rb_g_b, rb_s_b, wp12, accS, N);

    // resA = relu(spconv(in, rb_b, w2) + b2)
    hipMemsetAsync(accT, 0, accBytes, stream);
    spconv16<<<cgrid, 256, 0, stream>>>(inbf, rb_g_b, rb_s_b, wp2, accT, N);
    bias_relu_bf16<<<eblk, 256, 0, stream>>>(accT, b2, actb, total4);

    // resA = relu(spconv(resA, rb_a, w3) + b3) -> raw acc in accT
    hipMemsetAsync(accT, 0, accBytes, stream);
    spconv16<<<cgrid, 256, 0, stream>>>(actb, rb_g_a, rb_s_a, wp3, accT, N);

    // merge + logits
    merge_logits<<<(N + 255) / 256, 256, 0, stream>>>(accS, b1_2, accT, b3,
                                                      wl, bl, (float*)d_out, N, NC);
}
